// Encoder_82368882803150
// MI455X (gfx1250) — compile-verified, matrix operand order checked by reading, not run on hardware
//
#include <hip/hip_runtime.h>
#include <hip/hip_bf16.h>
#include <math.h>

// ---------------------------------------------------------------------------
// CDNA5 (gfx1250) WMMA-based Encoder:
//   embed-gather + LSTM(256x128, H=512) fused per-step with
//   V_WMMA_F32_16X16X32_BF16, bf16 operands / f32 accumulate.
// ---------------------------------------------------------------------------

typedef __bf16 bf16_t;
typedef __attribute__((ext_vector_type(16))) __bf16 v16bf;
typedef __attribute__((ext_vector_type(8)))  __bf16 v8bf;
typedef __attribute__((ext_vector_type(8)))  float  v8f;

#define KDIM   512     // E == H == 512
#define GATES  2048    // 4*H
#define NSENT  256     // N
#define SEQL   128     // L
#define BATCH  64
#define NJ     4       // n-1
#define VOCAB  32000
#define KVN    512

// Build a 16x32 bf16 WMMA fragment for one lane.
// 16-bit A/B layout (ISA 7.12.2): lanes 0-15 hold K = [k, k+8) and [k+16, k+24),
// lanes 16-31 hold the +8-shifted halves.  Rows are 512 bf16 -> every chunk is
// a 16-byte-aligned b128 load.
static __device__ __forceinline__ v16bf frag_cat(v8bf lo, v8bf hi) {
  return __builtin_shufflevector(lo, hi, 0,1,2,3,4,5,6,7,8,9,10,11,12,13,14,15);
}
static __device__ __forceinline__ v16bf load_frag(const bf16_t* __restrict__ row, int baseK) {
  v8bf lo = *(const v8bf*)(row + baseK);
  v8bf hi = *(const v8bf*)(row + baseK + 16);
  return frag_cat(lo, hi);
}
static __device__ __forceinline__ v8f wmma_bf16(v16bf a, v16bf b, v8f c) {
  // (neg_a, A, neg_b, B, c_mod, C, reuse_a, reuse_b)
  return __builtin_amdgcn_wmma_f32_16x16x32_bf16(false, a, false, b, (short)0, c, false, false);
}
static __device__ __forceinline__ float sigm(float x) { return 1.0f / (1.0f + __expf(-x)); }

// ---------------------------------------------------------------------------
// Prep kernels
// ---------------------------------------------------------------------------
__global__ void k_cvt_bf16(const float* __restrict__ src, bf16_t* __restrict__ dst, int n) {
  int i = blockIdx.x * blockDim.x + threadIdx.x;
  if (i < n) dst[i] = (bf16_t)src[i];
}

__global__ void k_invperm(const int* __restrict__ lst, int* __restrict__ inv, int n) {
  int i = blockIdx.x * blockDim.x + threadIdx.x;
  if (i < n) inv[lst[i]] = i;
}

// keys_out[k][e] = emb[keys[k][0]][e] + emb[keys[k][1]][e]
__global__ void k_keys(const int* __restrict__ keys, const float* __restrict__ emb,
                       float* __restrict__ out) {
  int k = blockIdx.x, e = threadIdx.x;
  int t0 = keys[k * 2 + 0], t1 = keys[k * 2 + 1];
  out[(size_t)k * KDIM + e] = emb[(size_t)t0 * KDIM + e] + emb[(size_t)t1 * KDIM + e];
}

// rowsum[n][e] = sum_t emb[tok[n][t]][e]
__global__ void k_rowsum(const int* __restrict__ tok, const float* __restrict__ emb,
                         float* __restrict__ out) {
  int n = blockIdx.x, e = threadIdx.x;
  float s = 0.f;
  for (int t = 0; t < SEQL; ++t)
    s += emb[(size_t)tok[n * SEQL + t] * KDIM + e];
  out[(size_t)n * KDIM + e] = s;
}

// cosine-vs-last + softmax over the 4 chunk positions -> weight[b][j]
__global__ void k_weight(const float* __restrict__ rowsum, const int* __restrict__ inv,
                         float* __restrict__ weight) {
  __shared__ float red[256];
  int b = blockIdx.x, tid = threadIdx.x;
  const float* s3 = rowsum + (size_t)inv[b * NJ + 3] * KDIM;
  float dots[NJ], qs[NJ];
  for (int j = 0; j < NJ; ++j) {
    const float* sj = rowsum + (size_t)inv[b * NJ + j] * KDIM;
    float d = 0.f, q = 0.f;
    for (int e = tid; e < KDIM; e += 256) { float v = sj[e]; d += v * s3[e]; q += v * v; }
    red[tid] = d; __syncthreads();
    for (int o = 128; o > 0; o >>= 1) { if (tid < o) red[tid] += red[tid + o]; __syncthreads(); }
    dots[j] = red[0]; __syncthreads();
    red[tid] = q; __syncthreads();
    for (int o = 128; o > 0; o >>= 1) { if (tid < o) red[tid] += red[tid + o]; __syncthreads(); }
    qs[j] = red[0]; __syncthreads();
  }
  if (tid == 0) {
    float n3 = fmaxf(sqrtf(qs[3]), 1e-8f);
    float c[NJ], mx = -1e30f;
    for (int j = 0; j < NJ; ++j) {
      c[j] = dots[j] / (fmaxf(sqrtf(qs[j]), 1e-8f) * n3);
      mx = fmaxf(mx, c[j]);
    }
    float se = 0.f;
    for (int j = 0; j < NJ; ++j) { c[j] = __expf(c[j] - mx); se += c[j]; }
    for (int j = 0; j < NJ; ++j) weight[b * NJ + j] = c[j] / se;
  }
}

// ---------------------------------------------------------------------------
// Fused encoder LSTM step: gates = embed_t @ W_ih^T + h @ W_hh^T + b
// One wave = one 16(batch) x 16(hidden) tile, 4 gate accumulators,
// 8 WMMAs per K-chunk.  512 waves/launch, 128 sequential launches.
// ---------------------------------------------------------------------------
__global__ __launch_bounds__(128)
void k_lstm_step_enc(const bf16_t* __restrict__ emb_bf, const int* __restrict__ tok,
                     const int* __restrict__ lens, int t,
                     const bf16_t* __restrict__ h_prev, bf16_t* __restrict__ h_next,
                     float* __restrict__ c,
                     const bf16_t* __restrict__ Wih, const bf16_t* __restrict__ Whh,
                     const float* __restrict__ bias,
                     float* __restrict__ enc_out) {
  int wid  = blockIdx.x * 4 + (threadIdx.x >> 5);
  int lane = threadIdx.x & 31;
  int mb = wid & 15;                 // 16 batch tiles (256 rows)
  int hb = wid >> 4;                 // 32 hidden tiles (512 cols)
  int mrow = mb * 16 + (lane & 15);  // A row this lane feeds
  int hcol = hb * 16 + (lane & 15);  // B row (gate column within H)
  int kof  = (lane >> 4) * 8;        // K half-select per 16-bit layout

  const bf16_t* arow_x = emb_bf + (size_t)tok[mrow * SEQL + t] * KDIM;
  const bf16_t* arow_h = h_prev + (size_t)mrow * KDIM;

  v8f acc[4] = {};                   // i, f, g, o accumulators (f32)
#pragma unroll 4
  for (int kk = 0; kk < KDIM; kk += 32) {
    int bK = kk + kof;
    v16bf ax = load_frag(arow_x, bK);
    v16bf ah = load_frag(arow_h, bK);
    __builtin_prefetch(arow_x + bK + 64, 0, 1);
#pragma unroll
    for (int g = 0; g < 4; ++g) {
      const bf16_t* wi = Wih + (size_t)(g * KDIM + hcol) * KDIM;
      const bf16_t* wh = Whh + (size_t)(g * KDIM + hcol) * KDIM;
      acc[g] = wmma_bf16(ax, load_frag(wi, bK), acc[g]);
      acc[g] = wmma_bf16(ah, load_frag(wh, bK), acc[g]);
    }
  }

  // D-matrix layout: VGPR r -> M = r + 8*(lane>=16), N = lane%16
  int hidx  = hcol;
  int mbase = mb * 16 + 8 * (lane >> 4);
  float bi = bias[0 * KDIM + hidx], bf = bias[1 * KDIM + hidx];
  float bg = bias[2 * KDIM + hidx], bo = bias[3 * KDIM + hidx];
#pragma unroll
  for (int r = 0; r < 8; ++r) {
    int m = mbase + r;
    float si = sigm(acc[0][r] + bi);
    float sf = sigm(acc[1][r] + bf);
    float tg = tanhf(acc[2][r] + bg);
    float so = sigm(acc[3][r] + bo);
    size_t ci = (size_t)m * KDIM + hidx;
    float c_old = c[ci];
    float c_new = sf * c_old + si * tg;
    float h_new = so * tanhf(c_new);
    bool  msk   = t < lens[m];
    c[ci] = msk ? c_new : c_old;
    float h_old = (float)h_prev[ci];
    h_next[ci]  = (bf16_t)(msk ? h_new : h_old);
    enc_out[((size_t)m * SEQL + t) * KDIM + hidx] = msk ? h_new : 0.0f;
  }
}

// ---------------------------------------------------------------------------
// Generic 256x512x2048 WMMA GEMM: out[r][n] = bias[n] + A[r] . W[n]
// (input projections of the two small LSTMs)
// ---------------------------------------------------------------------------
__global__ __launch_bounds__(128)
void k_gemm_xg(const bf16_t* __restrict__ A, const bf16_t* __restrict__ W,
               const float* __restrict__ bias, float* __restrict__ out) {
  int wid  = blockIdx.x * 4 + (threadIdx.x >> 5);
  int lane = threadIdx.x & 31;
  int mb = wid & 15;                 // 16 row tiles (256 rows)
  int nb = wid >> 4;                 // 128 col tiles (2048 cols)
  const bf16_t* arow = A + (size_t)(mb * 16 + (lane & 15)) * KDIM;
  const bf16_t* wrow = W + (size_t)(nb * 16 + (lane & 15)) * KDIM;
  int kof = (lane >> 4) * 8;
  v8f acc = {};
#pragma unroll 4
  for (int kk = 0; kk < KDIM; kk += 32)
    acc = wmma_bf16(load_frag(arow, kk + kof), load_frag(wrow, kk + kof), acc);
  int n = nb * 16 + (lane & 15);
  int mbase = mb * 16 + 8 * (lane >> 4);
  float bv = bias[n];
#pragma unroll
  for (int r = 0; r < 8; ++r)
    out[(size_t)(mbase + r) * GATES + n] = acc[r] + bv;
}

// ---------------------------------------------------------------------------
// Small LSTM step (M=64, all-true mask), xg precomputed.
// seq_out[b*4 + t][h] = h_new.
// ---------------------------------------------------------------------------
__global__ __launch_bounds__(128)
void k_lstm_step_small(const float* __restrict__ xg, int t,
                       const bf16_t* __restrict__ h_prev, bf16_t* __restrict__ h_next,
                       float* __restrict__ c, const bf16_t* __restrict__ Whh,
                       float* __restrict__ seq_out) {
  int wid  = blockIdx.x * 4 + (threadIdx.x >> 5);
  int lane = threadIdx.x & 31;
  int mb = wid & 3;                  // 4 row tiles (64 rows)
  int hb = wid >> 2;                 // 32 hidden tiles
  const bf16_t* arow = h_prev + (size_t)(mb * 16 + (lane & 15)) * KDIM;
  int hcol = hb * 16 + (lane & 15);
  int kof  = (lane >> 4) * 8;
  v8f acc[4] = {};
#pragma unroll 4
  for (int kk = 0; kk < KDIM; kk += 32) {
    v16bf ah = load_frag(arow, kk + kof);
#pragma unroll
    for (int g = 0; g < 4; ++g)
      acc[g] = wmma_bf16(ah, load_frag(Whh + (size_t)(g * KDIM + hcol) * KDIM, kk + kof), acc[g]);
  }
  int mbase = mb * 16 + 8 * (lane >> 4);
#pragma unroll
  for (int r = 0; r < 8; ++r) {
    int m = mbase + r;
    const float* xr = xg + (size_t)(m * NJ + t) * GATES;
    float si = sigm(acc[0][r] + xr[0 * KDIM + hcol]);
    float sf = sigm(acc[1][r] + xr[1 * KDIM + hcol]);
    float tg = tanhf(acc[2][r] + xr[2 * KDIM + hcol]);
    float so = sigm(acc[3][r] + xr[3 * KDIM + hcol]);
    size_t ci = (size_t)m * KDIM + hcol;
    float c_new = sf * c[ci] + si * tg;
    float h_new = so * tanhf(c_new);
    c[ci] = c_new;
    h_next[ci] = (bf16_t)h_new;
    seq_out[(size_t)(m * NJ + t) * KDIM + hcol] = h_new;
  }
}

// Permuted row copies feeding the small LSTMs.
__global__ void k_perm_bf16(const bf16_t* __restrict__ src, const int* __restrict__ inv,
                            bf16_t* __restrict__ dst) {
  int r = blockIdx.x, k = threadIdx.x;
  dst[(size_t)r * KDIM + k] = src[(size_t)inv[r] * KDIM + k];
}
__global__ void k_perm_f32_bf16(const float* __restrict__ src, const int* __restrict__ inv,
                                bf16_t* __restrict__ dst) {
  int r = blockIdx.x, k = threadIdx.x;
  dst[(size_t)r * KDIM + k] = (bf16_t)src[(size_t)inv[r] * KDIM + k];
}

// out[b][h] = sum_j seq[b*4+j][h] * weight[b*4+j]
__global__ void k_wsum(const float* __restrict__ seq, const float* __restrict__ weight,
                       float* __restrict__ out) {
  int b = blockIdx.x, h = threadIdx.x;
  float s = 0.f;
  for (int j = 0; j < NJ; ++j)
    s += seq[(size_t)(b * NJ + j) * KDIM + h] * weight[b * NJ + j];
  out[(size_t)b * KDIM + h] = s;
}

// ---------------------------------------------------------------------------
extern "C" void kernel_launch(void* const* d_in, const int* in_sizes, int n_in,
                              void* d_out, int out_size, void* d_ws, size_t ws_size,
                              hipStream_t stream) {
  const int*   tok   = (const int*)d_in[0];
  const int*   lens  = (const int*)d_in[1];
  const int*   keys  = (const int*)d_in[2];
  const int*   lst   = (const int*)d_in[3];
  const float* emb   = (const float*)d_in[4];
  const float* Wih   = (const float*)d_in[5];
  const float* Whh   = (const float*)d_in[6];
  const float* b0    = (const float*)d_in[7];
  const float* Wih_h = (const float*)d_in[8];
  const float* Whh_h = (const float*)d_in[9];
  const float* b_h   = (const float*)d_in[10];
  const float* Wih_c = (const float*)d_in[11];
  const float* Whh_c = (const float*)d_in[12];
  const float* b_c   = (const float*)d_in[13];

  float* out_enc  = (float*)d_out;                         // 256*128*512
  float* out_h    = out_enc + (size_t)NSENT * SEQL * KDIM; // 64*512
  float* out_c    = out_h + (size_t)BATCH * KDIM;          // 64*512
  float* out_keys = out_c + (size_t)BATCH * KDIM;          // 512*512

  // ---- workspace carve ----
  char* wsp = (char*)d_ws;
  auto alloc = [&](size_t bytes) -> void* {
    void* p = wsp;
    wsp += (bytes + 255) & ~(size_t)255;
    return p;
  };
  bf16_t* emb_bf  = (bf16_t*)alloc((size_t)VOCAB * KDIM * 2);
  bf16_t* wih_bf  = (bf16_t*)alloc((size_t)GATES * KDIM * 2);
  bf16_t* whh_bf  = (bf16_t*)alloc((size_t)GATES * KDIM * 2);
  bf16_t* wihh_bf = (bf16_t*)alloc((size_t)GATES * KDIM * 2);
  bf16_t* whhh_bf = (bf16_t*)alloc((size_t)GATES * KDIM * 2);
  bf16_t* wihc_bf = (bf16_t*)alloc((size_t)GATES * KDIM * 2);
  bf16_t* whhc_bf = (bf16_t*)alloc((size_t)GATES * KDIM * 2);
  bf16_t* hA      = (bf16_t*)alloc((size_t)NSENT * KDIM * 2);
  bf16_t* hB      = (bf16_t*)alloc((size_t)NSENT * KDIM * 2);
  float*  c1      = (float*)alloc((size_t)NSENT * KDIM * 4);
  int*    inv     = (int*)alloc(NSENT * 4);
  bf16_t* hin2    = (bf16_t*)alloc((size_t)NSENT * KDIM * 2);
  bf16_t* cin3    = (bf16_t*)alloc((size_t)NSENT * KDIM * 2);
  float*  xg2     = (float*)alloc((size_t)NSENT * GATES * 4);
  float*  xg3     = (float*)alloc((size_t)NSENT * GATES * 4);
  bf16_t* h2A     = (bf16_t*)alloc((size_t)BATCH * KDIM * 2);
  bf16_t* h2B     = (bf16_t*)alloc((size_t)BATCH * KDIM * 2);
  bf16_t* h3A     = (bf16_t*)alloc((size_t)BATCH * KDIM * 2);
  bf16_t* h3B     = (bf16_t*)alloc((size_t)BATCH * KDIM * 2);
  float*  c2      = (float*)alloc((size_t)BATCH * KDIM * 4);
  float*  c3      = (float*)alloc((size_t)BATCH * KDIM * 4);
  float*  hseq    = (float*)alloc((size_t)BATCH * NJ * KDIM * 4);
  float*  cseq    = (float*)alloc((size_t)BATCH * NJ * KDIM * 4);
  float*  rowsum  = (float*)alloc((size_t)NSENT * KDIM * 4);
  float*  weight  = (float*)alloc((size_t)BATCH * NJ * 4);

  // ---- one-time conversions / prep ----
  const int WN = GATES * KDIM;
  k_cvt_bf16<<<(VOCAB * KDIM + 255) / 256, 256, 0, stream>>>(emb, emb_bf, VOCAB * KDIM);
  k_cvt_bf16<<<(WN + 255) / 256, 256, 0, stream>>>(Wih, wih_bf, WN);
  k_cvt_bf16<<<(WN + 255) / 256, 256, 0, stream>>>(Whh, whh_bf, WN);
  k_cvt_bf16<<<(WN + 255) / 256, 256, 0, stream>>>(Wih_h, wihh_bf, WN);
  k_cvt_bf16<<<(WN + 255) / 256, 256, 0, stream>>>(Whh_h, whhh_bf, WN);
  k_cvt_bf16<<<(WN + 255) / 256, 256, 0, stream>>>(Wih_c, wihc_bf, WN);
  k_cvt_bf16<<<(WN + 255) / 256, 256, 0, stream>>>(Whh_c, whhc_bf, WN);
  k_invperm<<<1, 256, 0, stream>>>(lst, inv, NSENT);
  k_keys<<<KVN, KDIM, 0, stream>>>(keys, emb, out_keys);
  k_rowsum<<<NSENT, KDIM, 0, stream>>>(tok, emb, rowsum);
  k_weight<<<BATCH, 256, 0, stream>>>(rowsum, inv, weight);

  hipMemsetAsync(hA, 0, (size_t)NSENT * KDIM * 2, stream);
  hipMemsetAsync(c1, 0, (size_t)NSENT * KDIM * 4, stream);

  // ---- encoder LSTM: 128 fused WMMA steps, ping-pong h ----
  bf16_t* hp = hA; bf16_t* hn = hB;
  for (int t = 0; t < SEQL; ++t) {
    k_lstm_step_enc<<<128, 128, 0, stream>>>(emb_bf, tok, lens, t, hp, hn, c1,
                                             wih_bf, whh_bf, b0, out_enc);
    bf16_t* tmp = hp; hp = hn; hn = tmp;
  }
  // hp now holds hT (bf16), c1 holds cT (f32)

  // ---- small LSTMs over the permuted final states ----
  k_perm_bf16<<<NSENT, KDIM, 0, stream>>>(hp, inv, hin2);
  k_perm_f32_bf16<<<NSENT, KDIM, 0, stream>>>(c1, inv, cin3);
  k_gemm_xg<<<512, 128, 0, stream>>>(hin2, wihh_bf, b_h, xg2);
  k_gemm_xg<<<512, 128, 0, stream>>>(cin3, wihc_bf, b_c, xg3);

  hipMemsetAsync(h2A, 0, (size_t)BATCH * KDIM * 2, stream);
  hipMemsetAsync(h3A, 0, (size_t)BATCH * KDIM * 2, stream);
  hipMemsetAsync(c2, 0, (size_t)BATCH * KDIM * 4, stream);
  hipMemsetAsync(c3, 0, (size_t)BATCH * KDIM * 4, stream);

  bf16_t* h2p = h2A; bf16_t* h2n = h2B;
  bf16_t* h3p = h3A; bf16_t* h3n = h3B;
  for (int j = 0; j < NJ; ++j) {
    k_lstm_step_small<<<32, 128, 0, stream>>>(xg2, j, h2p, h2n, c2, whhh_bf, hseq);
    k_lstm_step_small<<<32, 128, 0, stream>>>(xg3, j, h3p, h3n, c3, whhc_bf, cseq);
    bf16_t* t2 = h2p; h2p = h2n; h2n = t2;
    bf16_t* t3 = h3p; h3p = h3n; h3n = t3;
  }

  // ---- attention-weighted reductions ----
  k_wsum<<<BATCH, KDIM, 0, stream>>>(hseq, weight, out_h);
  k_wsum<<<BATCH, KDIM, 0, stream>>>(cseq, weight, out_c);

  (void)in_sizes; (void)n_in; (void)out_size; (void)ws_size;
}
